// DABlock_74972949119409
// MI455X (gfx1250) — matrix operand order repaired and checked by reading, not compiled, not run on hardware
//
#include <hip/hip_runtime.h>
#include <math.h>

typedef __attribute__((ext_vector_type(16))) _Float16 v16h;
typedef __attribute__((ext_vector_type(8)))  _Float16 v8h;
typedef __attribute__((ext_vector_type(8)))  float    v8f;
typedef __attribute__((ext_vector_type(4)))  int      v4i;

#define BATCH 4
#define CH    512
#define HW    1024
#define NHEAD 8
#define HC    64
#define LDS_PAD 8          // halves; 16B row padding to rotate LDS banks

#if __has_builtin(__builtin_amdgcn_global_load_async_to_lds_b128) && \
    __has_builtin(__builtin_amdgcn_s_wait_asynccnt)
#define HAVE_ASYNC_LDS 1
typedef __attribute__((address_space(1))) v4i* as1_v4i;
typedef __attribute__((address_space(3))) v4i* as3_v4i;
#else
#define HAVE_ASYNC_LDS 0
#endif

// ---------------------------------------------------------------- utilities

__global__ void cast_f32_f16(const float* __restrict__ src, _Float16* __restrict__ dst, int n) {
    int i = blockIdx.x * blockDim.x + threadIdx.x;
    if (i < n) dst[i] = (_Float16)src[i];
}

// xT[b][p][c] = (f16) x[b][c][p]
__global__ void transpose_cast(const float* __restrict__ src, _Float16* __restrict__ dst) {
    int i = blockIdx.x * blockDim.x + threadIdx.x;   // b*CH*HW total
    int p = i & (HW - 1);
    int bc = i >> 10;
    int c = bc & (CH - 1);
    int b = bc >> 9;
    dst[((size_t)b * HW + p) * CH + c] = (_Float16)src[i];
}

// ------------------------------------------------- WMMA projection GEMMs
// out[b][o][p] = sum_c W[o][c] * BT[b][p][c]  (+bias)
// Block: 256 thr = 8 waves; one 16-row weight tile staged in LDS (async),
// each wave register-blocks 4 adjacent 16-col N tiles; B fragments are
// software-pipelined one K-step ahead (prefetch wraps on last step).
__global__ void gemm_proj(const _Float16* __restrict__ Wh,
                          const _Float16* __restrict__ BT,
                          const float* __restrict__ bias,
                          float* __restrict__ out_f32,
                          _Float16* __restrict__ out_tr,
                          _Float16* __restrict__ out_nt,
                          int mode) {
    __shared__ _Float16 atile[16 * (CH + LDS_PAD)];
    const int tid = threadIdx.x;
    const int wave = tid >> 5, lane = tid & 31;
    const int hl = lane >> 4, lm = lane & 15;
    const int m0 = blockIdx.y * 16;                 // o tile
    const int nbase = blockIdx.x * 512 + wave * 64; // p base for this wave
    const int b = blockIdx.z;

    // stage A tile (16 rows x 512 k, f16) into LDS: 1024 chunks of 8 halves
    const _Float16* asrc = Wh + (size_t)m0 * CH;
#pragma unroll
    for (int i = 0; i < 4; ++i) {
        int e = tid + i * 256;
        int row = e >> 6, col = (e & 63) * 8;
        const _Float16* g = asrc + (size_t)row * CH + col;
        _Float16* l = atile + row * (CH + LDS_PAD) + col;
#if HAVE_ASYNC_LDS
        __builtin_amdgcn_global_load_async_to_lds_b128(
            (as1_v4i)(uintptr_t)g, (as3_v4i)(uintptr_t)l, 0, 0);
#else
        *(v8h*)l = *(const v8h*)g;
#endif
    }
#if HAVE_ASYNC_LDS
    __builtin_amdgcn_s_wait_asynccnt(0);
#endif
    __syncthreads();

    const _Float16* brow[4];
#pragma unroll
    for (int t = 0; t < 4; ++t)
        brow[t] = BT + (size_t)b * HW * CH + (size_t)(nbase + t * 16 + lm) * CH + hl * 16;

    v8f acc[4] = {{}, {}, {}, {}};
    v16h bcur[4];
#pragma unroll
    for (int t = 0; t < 4; ++t) bcur[t] = *(const v16h*)(brow[t]);
    for (int k0 = 0; k0 < CH; k0 += 32) {
        const _Float16* ap = atile + lm * (CH + LDS_PAD) + k0 + hl * 16;
        v8h alo = *(const v8h*)ap;
        v8h ahi = *(const v8h*)(ap + 8);
        v16h a;
#pragma unroll
        for (int i = 0; i < 8; ++i) { a[i] = alo[i]; a[i + 8] = ahi[i]; }
        const int kn = (k0 + 32) & (CH - 1);      // wraps to 0 on last step
        v16h bnxt[4];
#pragma unroll
        for (int t = 0; t < 4; ++t) bnxt[t] = *(const v16h*)(brow[t] + kn);
#pragma unroll
        for (int t = 0; t < 4; ++t)
            acc[t] = __builtin_amdgcn_wmma_f32_16x16x32_f16(false, a, false, bcur[t],
                                                            (short)0, acc[t], false, false);
#pragma unroll
        for (int t = 0; t < 4; ++t) bcur[t] = bnxt[t];
    }
#pragma unroll
    for (int t = 0; t < 4; ++t) {
        const int p = nbase + t * 16 + lm;
        if (mode == 0) {
            v8h pk;
#pragma unroll
            for (int r = 0; r < 8; ++r) {
                int o = m0 + hl * 8 + r;
                float val = acc[t][r] + bias[o];
                out_f32[((size_t)b * CH + o) * HW + p] = val;
                pk[r] = (_Float16)val;
            }
            *(v8h*)(out_tr + ((size_t)b * HW + p) * CH + m0 + hl * 8) = pk;
        } else if (mode == 1) {
            v8h pk;
#pragma unroll
            for (int r = 0; r < 8; ++r) pk[r] = (_Float16)(acc[t][r] + bias[m0 + hl * 8 + r]);
            *(v8h*)(out_tr + ((size_t)b * HW + p) * CH + m0 + hl * 8) = pk;
        } else {
#pragma unroll
            for (int r = 0; r < 8; ++r) {
                int o = m0 + hl * 8 + r;
                out_nt[((size_t)b * CH + o) * HW + p] = (_Float16)(acc[t][r] + bias[o]);
            }
        }
    }
}

// ------------------------------------------------- offset network

__global__ void dwconv3x3(const float* __restrict__ q, const float* __restrict__ w,
                          const float* __restrict__ bias, float* __restrict__ out) {
    int tid = blockIdx.x * blockDim.x + threadIdx.x;   // BATCH*CH*HW
    int p = tid & (HW - 1);
    int bc = tid >> 10;
    int c = bc & (CH - 1);
    int y = p >> 5, x = p & 31;
    const float* img = q + (size_t)bc * HW;
    const float* wk = w + c * 9;
    float s = bias[c];
#pragma unroll
    for (int dy = -1; dy <= 1; ++dy)
#pragma unroll
        for (int dx = -1; dx <= 1; ++dx) {
            int yy = y + dy, xx = x + dx;
            if (yy >= 0 && yy < 32 && xx >= 0 && xx < 32)
                s += wk[(dy + 1) * 3 + (dx + 1)] * img[yy * 32 + xx];
        }
    out[tid] = s;
}

__global__ void ln_gelu(float* __restrict__ h, const float* __restrict__ lw,
                        const float* __restrict__ lb) {
    __shared__ float red[256];
    int bp = blockIdx.x;                    // b*HW + p
    int b = bp >> 10, p = bp & (HW - 1);
    float* base = h + (size_t)b * CH * HW + p;
    int t = threadIdx.x;
    float v0 = base[(size_t)t * HW];
    float v1 = base[(size_t)(t + 256) * HW];
    red[t] = v0 + v1; __syncthreads();
    for (int st = 128; st > 0; st >>= 1) {
        if (t < st) red[t] += red[t + st];
        __syncthreads();
    }
    float mu = red[0] * (1.0f / 512.0f); __syncthreads();
    float d0 = v0 - mu, d1 = v1 - mu;
    red[t] = d0 * d0 + d1 * d1; __syncthreads();
    for (int st = 128; st > 0; st >>= 1) {
        if (t < st) red[t] += red[t + st];
        __syncthreads();
    }
    float rstd = rsqrtf(red[0] * (1.0f / 512.0f) + 1e-5f);
    float y0 = d0 * rstd * lw[t] + lb[t];
    float y1 = d1 * rstd * lw[t + 256] + lb[t + 256];
    y0 = 0.5f * y0 * (1.0f + erff(y0 * 0.70710678f));
    y1 = 0.5f * y1 * (1.0f + erff(y1 * 0.70710678f));
    base[(size_t)t * HW] = y0;
    base[(size_t)(t + 256) * HW] = y1;
}

__global__ void pw_offset_pos(const float* __restrict__ h, const float* __restrict__ pw,
                              float* __restrict__ pos) {
    __shared__ float r0[128], r1[128];
    int bp = blockIdx.x;
    int b = bp >> 10, p = bp & (HW - 1);
    const float* base = h + (size_t)b * CH * HW + p;
    float s0 = 0.f, s1 = 0.f;
    for (int c = threadIdx.x; c < CH; c += 128) {
        float v = base[(size_t)c * HW];
        s0 += v * pw[c];
        s1 += v * pw[CH + c];
    }
    r0[threadIdx.x] = s0; r1[threadIdx.x] = s1; __syncthreads();
    for (int st = 64; st > 0; st >>= 1) {
        if (threadIdx.x < st) { r0[threadIdx.x] += r0[threadIdx.x + st];
                                r1[threadIdx.x] += r1[threadIdx.x + st]; }
        __syncthreads();
    }
    if (threadIdx.x == 0) {
        float offy = tanhf(r0[0]) * (2.0f / 32.0f);
        float offx = tanhf(r1[0]) * (2.0f / 32.0f);
        int py = p >> 5, px = p & 31;
        float ry = ((py + 0.5f) / 32.0f) * 2.0f - 1.0f;
        float rx = ((px + 0.5f) / 32.0f) * 2.0f - 1.0f;
        pos[(size_t)bp * 2 + 0] = offy + ry;
        pos[(size_t)bp * 2 + 1] = offx + rx;
    }
}

// deformable sampling: xsT[b][n][c] = bilinear(x[b][c], pos[b][n])
__global__ void sample_x(const float* __restrict__ x, const float* __restrict__ pos,
                         _Float16* __restrict__ xsT) {
    int bn = blockIdx.x;
    int b = bn >> 10;
    float py = pos[(size_t)bn * 2 + 0], px = pos[(size_t)bn * 2 + 1];
    float gx = (px + 1.0f) * 0.5f * 31.0f;
    float gy = (py + 1.0f) * 0.5f * 31.0f;
    float x0f = floorf(gx), y0f = floorf(gy);
    float wx1 = gx - x0f, wx0 = 1.0f - wx1;
    float wy1 = gy - y0f, wy0 = 1.0f - wy1;
    int x0 = (int)x0f, y0 = (int)y0f, x1 = x0 + 1, y1 = y0 + 1;
    float w00 = wx0 * wy0 * ((x0 >= 0 && x0 <= 31 && y0 >= 0 && y0 <= 31) ? 1.f : 0.f);
    float w10 = wx1 * wy0 * ((x1 >= 0 && x1 <= 31 && y0 >= 0 && y0 <= 31) ? 1.f : 0.f);
    float w01 = wx0 * wy1 * ((x0 >= 0 && x0 <= 31 && y1 >= 0 && y1 <= 31) ? 1.f : 0.f);
    float w11 = wx1 * wy1 * ((x1 >= 0 && x1 <= 31 && y1 >= 0 && y1 <= 31) ? 1.f : 0.f);
    int x0c = min(max(x0, 0), 31), x1c = min(max(x1, 0), 31);
    int y0c = min(max(y0, 0), 31), y1c = min(max(y1, 0), 31);
    int i00 = y0c * 32 + x0c, i10 = y0c * 32 + x1c;
    int i01 = y1c * 32 + x0c, i11 = y1c * 32 + x1c;
    const float* img = x + (size_t)b * CH * HW;
    _Float16* dst = xsT + (size_t)bn * CH;
    for (int c = threadIdx.x; c < CH; c += 128) {
        const float* chp = img + (size_t)c * HW;
        float v = w00 * chp[i00] + w10 * chp[i10] + w01 * chp[i01] + w11 * chp[i11];
        dst[c] = (_Float16)v;
    }
}

// ------------------------------------------------- attention GEMMs

// attn[bh][m][n] = scale * sum_c qT[b][m][h*64+c] * kT[b][n][h*64+c]
// one wave per block, 4 N-tiles register-blocked, B pipelined 1 step ahead
__global__ void gemm_attn(const _Float16* __restrict__ qT, const _Float16* __restrict__ kT,
                          float* __restrict__ attn) {
    const int lane = threadIdx.x;
    const int hl = lane >> 4, lm = lane & 15;
    const int n0 = blockIdx.x * 64, m0 = blockIdx.y * 16;
    const int bh = blockIdx.z, b = bh >> 3, h = bh & 7;
    const _Float16* A = qT + (size_t)b * HW * CH + h * HC;
    const _Float16* brow[4];
#pragma unroll
    for (int t = 0; t < 4; ++t)
        brow[t] = kT + (size_t)b * HW * CH + h * HC + (size_t)(n0 + t * 16 + lm) * CH + hl * 16;

    v8f acc[4] = {{}, {}, {}, {}};
    v16h bcur[4];
#pragma unroll
    for (int t = 0; t < 4; ++t) bcur[t] = *(const v16h*)(brow[t]);
#pragma unroll
    for (int k0 = 0; k0 < HC; k0 += 32) {
        const _Float16* ap = A + (size_t)(m0 + lm) * CH + k0 + hl * 16;
        v8h alo = *(const v8h*)ap;
        v8h ahi = *(const v8h*)(ap + 8);
        v16h a;
#pragma unroll
        for (int i = 0; i < 8; ++i) { a[i] = alo[i]; a[i + 8] = ahi[i]; }
        const int kn = (k0 + 32) & (HC - 1);
        v16h bnxt[4];
#pragma unroll
        for (int t = 0; t < 4; ++t) bnxt[t] = *(const v16h*)(brow[t] + kn);
#pragma unroll
        for (int t = 0; t < 4; ++t)
            acc[t] = __builtin_amdgcn_wmma_f32_16x16x32_f16(false, a, false, bcur[t],
                                                            (short)0, acc[t], false, false);
#pragma unroll
        for (int t = 0; t < 4; ++t) bcur[t] = bnxt[t];
    }
    const float scale = 0.125f;   // 64^-0.5
    float* obase = attn + (size_t)bh * HW * HW;
#pragma unroll
    for (int t = 0; t < 4; ++t)
#pragma unroll
        for (int r = 0; r < 8; ++r) {
            int m = m0 + hl * 8 + r;
            obase[(size_t)m * HW + n0 + t * 16 + lm] = acc[t][r] * scale;
        }
}

// fused rpe-bias (grid_sample over 63x63 table) + softmax over n
// writes f32 back into d_out and (optionally) an f16 copy for gemm_out
__global__ void bias_softmax(float* __restrict__ attn, const float* __restrict__ pos,
                             const float* __restrict__ rpe, _Float16* __restrict__ attn_h) {
    __shared__ float red[256];
    int blk = blockIdx.x;                 // bh*HW + m
    int m = blk & (HW - 1);
    int bh = blk >> 10;
    int b = bh >> 3, h = bh & 7;
    int my = m >> 5, mx = m & 31;
    float qy = ((my + 0.5f) / 32.f) * 2.f - 1.f;
    float qx = ((mx + 0.5f) / 32.f) * 2.f - 1.f;
    const float* rt = rpe + (size_t)h * 63 * 63;
    float* row = attn + (size_t)blk * HW;
    float vals[4];
    float mloc = -1e30f;
#pragma unroll
    for (int j = 0; j < 4; ++j) {
        int n = threadIdx.x + j * 256;
        float py = pos[((size_t)b * HW + n) * 2 + 0];
        float px = pos[((size_t)b * HW + n) * 2 + 1];
        float dy = (qy - py) * 0.5f, dx = (qx - px) * 0.5f;
        float gx = (dx + 1.f) * 31.f, gy = (dy + 1.f) * 31.f;
        float x0f = floorf(gx), y0f = floorf(gy);
        float wx1 = gx - x0f, wx0 = 1.f - wx1;
        float wy1 = gy - y0f, wy0 = 1.f - wy1;
        int x0 = (int)x0f, y0 = (int)y0f, x1 = x0 + 1, y1 = y0 + 1;
        float w00 = wx0 * wy0 * ((x0 >= 0 && x0 <= 62 && y0 >= 0 && y0 <= 62) ? 1.f : 0.f);
        float w10 = wx1 * wy0 * ((x1 >= 0 && x1 <= 62 && y0 >= 0 && y0 <= 62) ? 1.f : 0.f);
        float w01 = wx0 * wy1 * ((x0 >= 0 && x0 <= 62 && y1 >= 0 && y1 <= 62) ? 1.f : 0.f);
        float w11 = wx1 * wy1 * ((x1 >= 0 && x1 <= 62 && y1 >= 0 && y1 <= 62) ? 1.f : 0.f);
        int x0c = min(max(x0, 0), 62), x1c = min(max(x1, 0), 62);
        int y0c = min(max(y0, 0), 62), y1c = min(max(y1, 0), 62);
        float bias = w00 * rt[y0c * 63 + x0c] + w10 * rt[y0c * 63 + x1c]
                   + w01 * rt[y1c * 63 + x0c] + w11 * rt[y1c * 63 + x1c];
        float v = row[n] + bias;
        vals[j] = v;
        mloc = fmaxf(mloc, v);
    }
    red[threadIdx.x] = mloc; __syncthreads();
    for (int st = 128; st > 0; st >>= 1) {
        if (threadIdx.x < st) red[threadIdx.x] = fmaxf(red[threadIdx.x], red[threadIdx.x + st]);
        __syncthreads();
    }
    float rowmax = red[0]; __syncthreads();
    float s = 0.f;
#pragma unroll
    for (int j = 0; j < 4; ++j) { vals[j] = __expf(vals[j] - rowmax); s += vals[j]; }
    red[threadIdx.x] = s; __syncthreads();
    for (int st = 128; st > 0; st >>= 1) {
        if (threadIdx.x < st) red[threadIdx.x] += red[threadIdx.x + st];
        __syncthreads();
    }
    float inv = 1.0f / red[0];
    if (attn_h) {
        _Float16* hrow = attn_h + (size_t)blk * HW;
#pragma unroll
        for (int j = 0; j < 4; ++j) {
            float v = vals[j] * inv;
            row[threadIdx.x + j * 256] = v;
            hrow[threadIdx.x + j * 256] = (_Float16)v;
        }
    } else {
#pragma unroll
        for (int j = 0; j < 4; ++j) row[threadIdx.x + j * 256] = vals[j] * inv;
    }
}

// out[b][h*64+c][m] = gamma * sum_n attn[bh][m][n] * v[b][h*64+c][n] + x[...]
// f16-attn variant: contiguous v16h B loads, pipelined 1 K-step ahead
__global__ void gemm_out_h(const _Float16* __restrict__ vh, const _Float16* __restrict__ attn_h,
                           const float* __restrict__ x, const float* __restrict__ gamma,
                           float* __restrict__ out) {
    const int lane = threadIdx.x;
    const int hl = lane >> 4, lm = lane & 15;
    const int n0 = blockIdx.x * 64;      // m (spatial) tile group
    const int m0 = blockIdx.y * 16;      // c-within-head tile
    const int bh = blockIdx.z, b = bh >> 3, h = bh & 7;
    const _Float16* A = vh + ((size_t)b * CH + h * HC) * HW;
    const _Float16* brow[4];
#pragma unroll
    for (int t = 0; t < 4; ++t)
        brow[t] = attn_h + (size_t)bh * HW * HW + (size_t)(n0 + t * 16 + lm) * HW + hl * 16;

    v8f acc[4] = {{}, {}, {}, {}};
    v16h bcur[4];
#pragma unroll
    for (int t = 0; t < 4; ++t) bcur[t] = *(const v16h*)(brow[t]);
    for (int k0 = 0; k0 < HW; k0 += 32) {
        const _Float16* ap = A + (size_t)(m0 + lm) * HW + k0 + hl * 16;
        v8h alo = *(const v8h*)ap;
        v8h ahi = *(const v8h*)(ap + 8);
        v16h a;
#pragma unroll
        for (int i = 0; i < 8; ++i) { a[i] = alo[i]; a[i + 8] = ahi[i]; }
        const int kn = (k0 + 32) & (HW - 1);
        v16h bnxt[4];
#pragma unroll
        for (int t = 0; t < 4; ++t) bnxt[t] = *(const v16h*)(brow[t] + kn);
#pragma unroll
        for (int t = 0; t < 4; ++t)
            acc[t] = __builtin_amdgcn_wmma_f32_16x16x32_f16(false, a, false, bcur[t],
                                                            (short)0, acc[t], false, false);
#pragma unroll
        for (int t = 0; t < 4; ++t) bcur[t] = bnxt[t];
    }
    float g = gamma[0];
#pragma unroll
    for (int t = 0; t < 4; ++t)
#pragma unroll
        for (int r = 0; r < 8; ++r) {
            int cl = m0 + hl * 8 + r;
            size_t idx = ((size_t)b * CH + h * HC + cl) * HW + n0 + t * 16 + lm;
            out[idx] = g * acc[t][r] + x[idx];
        }
}

// fallback: read f32 attn from d_out and convert in-register
__global__ void gemm_out_f(const _Float16* __restrict__ vh, const float* __restrict__ attn,
                           const float* __restrict__ x, const float* __restrict__ gamma,
                           float* __restrict__ out) {
    const int lane = threadIdx.x;
    const int hl = lane >> 4, lm = lane & 15;
    const int n0 = blockIdx.x * 64;
    const int m0 = blockIdx.y * 16;
    const int bh = blockIdx.z, b = bh >> 3, h = bh & 7;
    const _Float16* A = vh + ((size_t)b * CH + h * HC) * HW;
    const float* Bt = attn + (size_t)bh * HW * HW;
    v8f acc[4] = {{}, {}, {}, {}};
    for (int k0 = 0; k0 < HW; k0 += 32) {
        const _Float16* ap = A + (size_t)(m0 + lm) * HW + k0 + hl * 16;
        v8h alo = *(const v8h*)ap;
        v8h ahi = *(const v8h*)(ap + 8);
        v16h a;
#pragma unroll
        for (int i = 0; i < 8; ++i) { a[i] = alo[i]; a[i + 8] = ahi[i]; }
#pragma unroll
        for (int t = 0; t < 4; ++t) {
            const float* bp = Bt + (size_t)(n0 + t * 16 + lm) * HW + k0 + hl * 16;
            v16h bm;
#pragma unroll
            for (int i = 0; i < 16; ++i) bm[i] = (_Float16)bp[i];
            acc[t] = __builtin_amdgcn_wmma_f32_16x16x32_f16(false, a, false, bm,
                                                            (short)0, acc[t], false, false);
        }
    }
    float g = gamma[0];
#pragma unroll
    for (int t = 0; t < 4; ++t)
#pragma unroll
        for (int r = 0; r < 8; ++r) {
            int cl = m0 + hl * 8 + r;
            size_t idx = ((size_t)b * CH + h * HC + cl) * HW + n0 + t * 16 + lm;
            out[idx] = g * acc[t][r] + x[idx];
        }
}

// ---------------------------------------------------------------- launcher

#define WS_QW   ((size_t)0)
#define WS_KW   ((size_t)524288)
#define WS_VW   ((size_t)1048576)
#define WS_XT   ((size_t)1572864)
#define WS_QT   ((size_t)5767168)
#define WS_QF   ((size_t)9961472)
#define WS_HB   ((size_t)18350080)
#define WS_POS  ((size_t)26738688)
#define WS_XST  ((size_t)26771456)
#define WS_KT   ((size_t)30965760)
#define WS_VH   ((size_t)35160064)
#define WS_AH   ((size_t)39354368)
#define WS_END_BIG   (WS_AH + (size_t)BATCH * NHEAD * HW * HW * 2)

extern "C" void kernel_launch(void* const* d_in, const int* in_sizes, int n_in,
                              void* d_out, int out_size, void* d_ws, size_t ws_size,
                              hipStream_t stream) {
    const float* x      = (const float*)d_in[0];
    const float* dw_w   = (const float*)d_in[1];
    const float* dw_b   = (const float*)d_in[2];
    const float* ln_w   = (const float*)d_in[3];
    const float* ln_b   = (const float*)d_in[4];
    const float* pw_w   = (const float*)d_in[5];
    const float* q_w    = (const float*)d_in[6];
    const float* q_b    = (const float*)d_in[7];
    const float* k_w    = (const float*)d_in[8];
    const float* k_b    = (const float*)d_in[9];
    const float* v_w    = (const float*)d_in[10];
    const float* v_b    = (const float*)d_in[11];
    const float* rpe    = (const float*)d_in[12];
    const float* gamma  = (const float*)d_in[13];

    float* out_f = (float*)d_out;                     // [B,C,H,W]
    float* attn  = out_f + (size_t)BATCH * CH * HW;   // [B*heads, HW, n]

    char* ws = (char*)d_ws;
    _Float16* qw_h = (_Float16*)(ws + WS_QW);
    _Float16* kw_h = (_Float16*)(ws + WS_KW);
    _Float16* vw_h = (_Float16*)(ws + WS_VW);
    _Float16* xT_h = (_Float16*)(ws + WS_XT);
    _Float16* qT_h = (_Float16*)(ws + WS_QT);
    float*    q_f  = (float*)   (ws + WS_QF);
    float*    hbuf = (float*)   (ws + WS_HB);
    float*    pos  = (float*)   (ws + WS_POS);
    _Float16* xsT  = (_Float16*)(ws + WS_XST);
    _Float16* kT_h = (_Float16*)(ws + WS_KT);
    _Float16* v_h  = (_Float16*)(ws + WS_VH);
    const bool use_h = (ws_size >= WS_END_BIG);
    _Float16* attn_h = use_h ? (_Float16*)(ws + WS_AH) : nullptr;

    const int WN = CH * CH;                 // 262144 weight elems
    cast_f32_f16<<<WN / 256, 256, 0, stream>>>(q_w, qw_h, WN);
    cast_f32_f16<<<WN / 256, 256, 0, stream>>>(k_w, kw_h, WN);
    cast_f32_f16<<<WN / 256, 256, 0, stream>>>(v_w, vw_h, WN);
    transpose_cast<<<(BATCH * CH * HW) / 256, 256, 0, stream>>>(x, xT_h);

    dim3 gproj(HW / 512, CH / 16, BATCH);   // (2, 32, 4) x 256 threads
    gemm_proj<<<gproj, 256, 0, stream>>>(qw_h, xT_h, q_b, q_f, qT_h, nullptr, 0);

    dwconv3x3<<<(BATCH * CH * HW) / 256, 256, 0, stream>>>(q_f, dw_w, dw_b, hbuf);
    ln_gelu<<<BATCH * HW, 256, 0, stream>>>(hbuf, ln_w, ln_b);
    pw_offset_pos<<<BATCH * HW, 128, 0, stream>>>(hbuf, pw_w, pos);
    sample_x<<<BATCH * HW, 128, 0, stream>>>(x, pos, xsT);

    gemm_proj<<<gproj, 256, 0, stream>>>(kw_h, xsT, k_b, nullptr, kT_h, nullptr, 1);
    gemm_proj<<<gproj, 256, 0, stream>>>(vw_h, xsT, v_b, nullptr, nullptr, v_h, 2);

    dim3 gattn(HW / 64, HW / 16, BATCH * NHEAD);  // (16, 64, 32), 1 wave/block
    gemm_attn<<<gattn, 32, 0, stream>>>(qT_h, kT_h, attn);

    bias_softmax<<<BATCH * NHEAD * HW, 256, 0, stream>>>(attn, pos, rpe, attn_h);

    dim3 gout(HW / 64, HC / 16, BATCH * NHEAD);   // (16, 4, 32), 1 wave/block
    if (use_h)
        gemm_out_h<<<gout, 32, 0, stream>>>(v_h, attn_h, x, gamma, out_f);
    else
        gemm_out_f<<<gout, 32, 0, stream>>>(v_h, attn, x, gamma, out_f);
}
